// MoE_28948079575212
// MI455X (gfx1250) — compile-verified, matrix operand order checked by reading, not run on hardware
//
#include <hip/hip_runtime.h>
#include <hip/hip_bf16.h>
#include <math.h>

// ---------------- problem constants (match reference) ----------------
#define NN      50000     // N_NODES
#define NE      800000    // N_EDGES
#define DIN     256
#define DOUT    256
#define NEXP    8
#define TOPK    3
#define LOSS_COEF 0.01f
#define EPS_LOSS  1e-10f

typedef __attribute__((ext_vector_type(16))) __bf16 v16bf;
typedef __attribute__((ext_vector_type(8)))  __bf16 v8bf;
typedef __attribute__((ext_vector_type(8)))  float  v8f;

// ---------------- workspace layout (bytes) ----------------
// agg    : NN*DIN f32         = 51,200,000
// gates  : NN*NEXP f32        =  1,600,000
// packedW: NEXP*DIN*DOUT bf16 =  1,048,576
// stats  : 16 f32 (importance[8], load[8])
#define WS_AGG    0
#define WS_GATES  51200000
#define WS_PW     52800000
#define WS_STATS  53848576

// scheduling hint helpers (pin 16 VMEM loads + 2 DS loads before 8 WMMAs)
__device__ __forceinline__ void sched_hint_kchunk() {
#if __has_builtin(__builtin_amdgcn_sched_group_barrier)
    __builtin_amdgcn_sched_group_barrier(0x020, 16, 0);  // 16 VMEM reads
    __builtin_amdgcn_sched_group_barrier(0x100, 2, 0);   // 2 DS reads
    __builtin_amdgcn_sched_group_barrier(0x008, 8, 0);   // 8 WMMA
#elif __has_builtin(__builtin_amdgcn_sched_barrier)
    __builtin_amdgcn_sched_barrier(0);
#endif
}

// =====================================================================
// 0) zero fill
// =====================================================================
__global__ void moe_zero_f32(float* __restrict__ p, int n) {
    int i = blockIdx.x * blockDim.x + threadIdx.x;
    int stride = gridDim.x * blockDim.x;
    for (; i < n; i += stride) p[i] = 0.0f;
}

// =====================================================================
// 1) scatter-add: agg[dst] += x[src]  (edge-parallel, float4 granules)
//    x and agg are both L2-resident (51 MB each vs 192 MB L2) ->
//    f32 global atomics resolve in L2 at high rate.
// =====================================================================
__global__ void moe_scatter_add(const float* __restrict__ x,
                                const int*   __restrict__ ei,
                                float*       __restrict__ agg) {
    long long tid = (long long)blockIdx.x * blockDim.x + threadIdx.x;
    int edge = (int)(tid >> 6);          // 64 threads per edge
    int c    = (int)(tid & 63) << 2;     // 4 floats per thread
    if (edge >= NE) return;
    int src = ei[edge];
    int dst = ei[NE + edge];
    float4 v = *(const float4*)(x + (long long)src * DIN + c);
    float* a = agg + (long long)dst * DIN + c;
    __hip_atomic_fetch_add(a + 0, v.x, __ATOMIC_RELAXED, __HIP_MEMORY_SCOPE_AGENT);
    __hip_atomic_fetch_add(a + 1, v.y, __ATOMIC_RELAXED, __HIP_MEMORY_SCOPE_AGENT);
    __hip_atomic_fetch_add(a + 2, v.z, __ATOMIC_RELAXED, __HIP_MEMORY_SCOPE_AGENT);
    __hip_atomic_fetch_add(a + 3, v.w, __ATOMIC_RELAXED, __HIP_MEMORY_SCOPE_AGENT);
}

// =====================================================================
// 2) repack W_exp (f32 [E][K=256][N=256] row-major) into exact WMMA
//    B-fragment order, bf16. Fragment f = ((e*8+kc)*16+ct) holds a
//    32x16 (KxN) tile; per lane 16 halves in the 16-bit operand layout:
//      N = ct*16 + (lane&15)
//      K = kc*32 + ( v<4 ? 2v+hi : 16+2(v-4)+hi ) + (lane&16 ? 8 : 0)
//    so the GEMM loads each fragment as pw + f*512 + lane*16 (2x b128).
// =====================================================================
__global__ void moe_repack_w(const float* __restrict__ W,
                             __bf16*      __restrict__ pw) {
    int idx = blockIdx.x * blockDim.x + threadIdx.x;
    if (idx >= NEXP * 8 * 16 * 512) return;
    int p    = idx & 15;
    int lane = (idx >> 4) & 31;
    int f    = idx >> 9;
    int ct   = f & 15;
    int kc   = (f >> 4) & 7;
    int e    = f >> 7;
    int v = p >> 1, hi = p & 1;
    int kofs = (v < 4 ? 2 * v + hi : 16 + 2 * (v - 4) + hi) + ((lane & 16) ? 8 : 0);
    int K = kc * 32 + kofs;
    int N = ct * 16 + (lane & 15);
    pw[idx] = (__bf16)W[((long long)e * DIN + K) * DOUT + N];
}

// =====================================================================
// 3) gate: one wave per node. logits = relu(agg @ Wg + bg),
//    top-3 of 8 -> softmax -> dense gates row; atomics for importance/load.
// =====================================================================
__global__ void moe_gate(const float* __restrict__ agg,
                         const float* __restrict__ Wg,
                         const float* __restrict__ bg,
                         float*       __restrict__ gates,
                         float*       __restrict__ stats) {
    int wave = threadIdx.x >> 5;
    int lane = threadIdx.x & 31;
    int n = blockIdx.x * 8 + wave;
    if (n >= NN) return;                       // uniform per wave

    float acc[8] = {0.f, 0.f, 0.f, 0.f, 0.f, 0.f, 0.f, 0.f};
    const float* row = agg + (long long)n * DIN;
#pragma unroll
    for (int i = 0; i < 8; ++i) {
        int d = i * 32 + lane;
        float xv = row[d];
        float4 w0 = *(const float4*)(Wg + d * 8);
        float4 w1 = *(const float4*)(Wg + d * 8 + 4);
        acc[0] += xv * w0.x; acc[1] += xv * w0.y;
        acc[2] += xv * w0.z; acc[3] += xv * w0.w;
        acc[4] += xv * w1.x; acc[5] += xv * w1.y;
        acc[6] += xv * w1.z; acc[7] += xv * w1.w;
    }
#pragma unroll
    for (int off = 16; off > 0; off >>= 1)
#pragma unroll
        for (int e = 0; e < 8; ++e)
            acc[e] += __shfl_xor(acc[e], off, 32);

    float lg[8];
#pragma unroll
    for (int e = 0; e < 8; ++e) lg[e] = fmaxf(acc[e] + bg[e], 0.0f);

    // top-3 (ties -> lowest index, matching lax.top_k)
    int ti[3]; float tv[3]; unsigned used = 0;
#pragma unroll
    for (int t = 0; t < 3; ++t) {
        float best = -INFINITY; int bi = 0;
#pragma unroll
        for (int e = 0; e < 8; ++e)
            if (!((used >> e) & 1u) && lg[e] > best) { best = lg[e]; bi = e; }
        ti[t] = bi; tv[t] = best; used |= 1u << bi;
    }
    float m  = tv[0];
    float e0 = __expf(tv[0] - m), e1 = __expf(tv[1] - m), e2 = __expf(tv[2] - m);
    float s  = e0 + e1 + e2;
    float g0 = e0 / s, g1 = e1 / s, g2 = e2 / s;

    if (lane == 0) {
        float out[8] = {0.f, 0.f, 0.f, 0.f, 0.f, 0.f, 0.f, 0.f};
        out[ti[0]] = g0; out[ti[1]] = g1; out[ti[2]] = g2;
#pragma unroll
        for (int e = 0; e < 8; ++e) gates[(long long)n * 8 + e] = out[e];
        __hip_atomic_fetch_add(stats + ti[0], g0, __ATOMIC_RELAXED, __HIP_MEMORY_SCOPE_AGENT);
        __hip_atomic_fetch_add(stats + ti[1], g1, __ATOMIC_RELAXED, __HIP_MEMORY_SCOPE_AGENT);
        __hip_atomic_fetch_add(stats + ti[2], g2, __ATOMIC_RELAXED, __HIP_MEMORY_SCOPE_AGENT);
        __hip_atomic_fetch_add(stats + 8 + ti[0], 1.0f, __ATOMIC_RELAXED, __HIP_MEMORY_SCOPE_AGENT);
        __hip_atomic_fetch_add(stats + 8 + ti[1], 1.0f, __ATOMIC_RELAXED, __HIP_MEMORY_SCOPE_AGENT);
        __hip_atomic_fetch_add(stats + 8 + ti[2], 1.0f, __ATOMIC_RELAXED, __HIP_MEMORY_SCOPE_AGENT);
    }
}

// =====================================================================
// 4) fused expert GEMM + gate combine, per-expert accumulators:
//      acc_e += A_kc @ B_{e,kc}          (pure WMMA inner loop, A shared)
//      y[m,n] = sum_e g[m,e] * (acc_e[m,n] + b_e[n])   (f32 epilogue)
//    Block = 128 thr (4 waves), 64 node rows; wave w owns row-tile w and
//    ALL 4 waves walk the 16 col-tiles together -> the block's live B
//    working set is one col-tile (64 fragments = 64 KB = WGP$ size),
//    re-hit 4x in-block. sched_group_barrier pins "16 loads -> 8 WMMA"
//    per k-chunk so all 8 B fragments stay live in distinct registers
//    and waits are graduated instead of wait-at-zero.
//    A staged in LDS as bf16, row stride 264 halves: 528 B row pitch,
//    (528/4) mod 64 = 4 -> 16 rows hit 16 distinct banks for b128 reads.
// =====================================================================
#define RB 64
#define ASTRIDE 264

__global__ __launch_bounds__(128, 1) void moe_expert_gemm(
        const float*  __restrict__ agg,
        const float*  __restrict__ gates,
        const __bf16* __restrict__ pw,
        const float*  __restrict__ bexp,
        float*        __restrict__ y) {
    __shared__ __bf16 sA[RB * ASTRIDE];
    __shared__ float  sG[RB * 8];

    int t = threadIdx.x;                   // 0..127
    long long rowBase = (long long)blockIdx.x * RB;

    // stage A tile (bf16) and gates into LDS
    for (int idx = t; idx < RB * DIN; idx += 128) {
        int r = idx >> 8, c = idx & 255;
        long long grow = rowBase + r;
        float v = (grow < NN) ? agg[grow * DIN + c] : 0.0f;
        sA[r * ASTRIDE + c] = (__bf16)v;
    }
    for (int idx = t; idx < RB * 8; idx += 128) {
        int r = idx >> 3, e = idx & 7;
        long long grow = rowBase + r;
        sG[idx] = (grow < NN) ? gates[grow * 8 + e] : 0.0f;
    }
    __syncthreads();

    int wave = t >> 5;                     // 0..3 == row-tile
    int lane = t & 31;
    int rt    = wave;
    int lrow  = lane & 15;                 // row (A) / col (B,D) within tile
    int khalf = (lane & 16) ? 8 : 0;       // upper-lane K offset

    const __bf16* pa = sA + (rt * 16 + lrow) * ASTRIDE + khalf;

    // A fragment loader (2x ds_load_b128 per k-chunk)
    auto loadA = [&](int kc) -> v16bf {
        v8bf lo = *(const v8bf*)(pa + kc * 32);        // K = kc*32 + {0..7}+khalf
        v8bf hi = *(const v8bf*)(pa + kc * 32 + 16);   // K = kc*32 + {16..23}+khalf
        v16bf a;
#pragma unroll
        for (int i = 0; i < 8; ++i) { a[i] = lo[i]; a[8 + i] = hi[i]; }
        return a;
    };

    // all 4 waves sweep col-tiles in the same order (shared B stream)
    for (int ctile = 0; ctile < 16; ++ctile) {
        const __bf16* pb = pw + (long long)ctile * 512 + lane * 16;

        // 8 per-expert f32 accumulators (64 VGPRs)
        v8f acc[NEXP];
#pragma unroll
        for (int e = 0; e < NEXP; ++e) acc[e] = (v8f){};

        v16bf av = loadA(0);

#pragma unroll
        for (int kc = 0; kc < 8; ++kc) {
            // ALL 8 B fragments live at once -> one 16-load clause
            v16bf bfr[NEXP];
#pragma unroll
            for (int e = 0; e < NEXP; ++e)
                bfr[e] = *(const v16bf*)(pb + (long long)(e * 8 + kc) * 8192);

            // prefetch next k-chunk's A fragment during the WMMA body
            v16bf anext = av;
            if (kc < 7) anext = loadA(kc + 1);

#pragma unroll
            for (int e = 0; e < NEXP; ++e)
                acc[e] = __builtin_amdgcn_wmma_f32_16x16x32_bf16(
                        false, av, false, bfr[e], (short)0, acc[e], false, false);
            av = anext;

            sched_hint_kchunk();           // 16 VMEM -> 2 DS -> 8 WMMA
        }

        // epilogue: gate-combine in f32, add gated bias, store
        int col = ctile * 16 + lrow;
        float be[NEXP];
#pragma unroll
        for (int e = 0; e < NEXP; ++e) be[e] = bexp[e * DOUT + col];

#pragma unroll
        for (int j = 0; j < 8; ++j) {
            int M = rt * 16 + j + ((lane & 16) ? 8 : 0);
            long long grow = rowBase + M;
            if (grow < NN) {
                float4 ga = *(const float4*)(sG + M * 8);
                float4 gb = *(const float4*)(sG + M * 8 + 4);
                float val = 0.0f;
                val += ga.x * (acc[0][j] + be[0]);
                val += ga.y * (acc[1][j] + be[1]);
                val += ga.z * (acc[2][j] + be[2]);
                val += ga.w * (acc[3][j] + be[3]);
                val += gb.x * (acc[4][j] + be[4]);
                val += gb.y * (acc[5][j] + be[5]);
                val += gb.z * (acc[6][j] + be[6]);
                val += gb.w * (acc[7][j] + be[7]);
                y[grow * DOUT + col] = val;
            }
        }
    }
}

// =====================================================================
// 5) aux loss: 0.01*(cv2(importance)+cv2(load)), var with ddof=1 over 8
// =====================================================================
__global__ void moe_loss(const float* __restrict__ stats, float* __restrict__ out) {
    if (threadIdx.x == 0 && blockIdx.x == 0) {
        float cv[2];
        for (int h = 0; h < 2; ++h) {
            const float* v = stats + h * 8;
            float s = 0.0f;
            for (int e = 0; e < 8; ++e) s += v[e];
            float mean = s * 0.125f;
            float q = 0.0f;
            for (int e = 0; e < 8; ++e) { float d = v[e] - mean; q += d * d; }
            float var = q / 7.0f;
            cv[h] = var / (mean * mean + EPS_LOSS);
        }
        out[(long long)NN * DOUT] = LOSS_COEF * (cv[0] + cv[1]);
    }
}

// =====================================================================
extern "C" void kernel_launch(void* const* d_in, const int* in_sizes, int n_in,
                              void* d_out, int out_size, void* d_ws, size_t ws_size,
                              hipStream_t stream) {
    const float* x  = (const float*)d_in[0];
    const int*   ei = (const int*)  d_in[1];
    const float* Wg = (const float*)d_in[2];
    const float* bg = (const float*)d_in[3];
    const float* We = (const float*)d_in[4];
    const float* be = (const float*)d_in[5];
    float* y = (float*)d_out;

    char* ws = (char*)d_ws;
    float*  agg   = (float*) (ws + WS_AGG);
    float*  gates = (float*) (ws + WS_GATES);
    __bf16* pw    = (__bf16*)(ws + WS_PW);
    float*  stats = (float*) (ws + WS_STATS);

    moe_zero_f32<<<2048, 256, 0, stream>>>(agg, NN * DIN);
    moe_zero_f32<<<1, 32, 0, stream>>>(stats, 16);
    moe_repack_w<<<(NEXP * 8 * 16 * 512 + 255) / 256, 256, 0, stream>>>(We, pw);
    moe_scatter_add<<<(int)(((long long)NE * 64) / 256), 256, 0, stream>>>(x, ei, agg);
    moe_gate<<<(NN + 7) / 8, 256, 0, stream>>>(agg, Wg, bg, gates, stats);
    moe_expert_gemm<<<(NN + RB - 1) / RB, 128, 0, stream>>>(agg, gates, pw, be, y);
    moe_loss<<<1, 32, 0, stream>>>(stats, y);
}
